// MultiHeadAttention_39273180955128
// MI455X (gfx1250) — compile-verified, matrix operand order checked by reading, not run on hardware
//
#include <hip/hip_runtime.h>
#include <hip/hip_bf16.h>

typedef __attribute__((ext_vector_type(16))) __bf16 v16bf;
typedef __attribute__((ext_vector_type(8)))  __bf16 v8bf;
typedef __attribute__((ext_vector_type(8)))  float  v8f;

#define HIDDEN   1024
#define HEADS    16
#define HEAD_DIM 64
#define BATCH    2
#define SEQ      2048
#define NTOK     (BATCH * SEQ)   /* 4096 */

#define WMMA_BF16(a, b, c) \
  __builtin_amdgcn_wmma_f32_16x16x32_bf16(false, (a), false, (b), (short)0, (c), false, false)

// Load a 16x32 bf16 A-fragment (or symmetric B-fragment) row slice.
// rowp points to the start of this lane's row (M = lane%16 for A, N = lane%16 for B),
// k0 is the 32-wide K-chunk base. Per ISA 7.12.2: element j -> K = (j&7) + (j>=8?16:0) + half*8,
// i.e. two contiguous 16-byte runs.
static __device__ __forceinline__ v16bf load_frag(const __bf16* rowp, int k0, int half) {
  const v8bf lo = *reinterpret_cast<const v8bf*>(rowp + k0 + half * 8);
  const v8bf hi = *reinterpret_cast<const v8bf*>(rowp + k0 + 16 + half * 8);
  v16bf a;
#pragma unroll
  for (int i = 0; i < 8; ++i) { a[i] = lo[i]; a[i + 8] = hi[i]; }
  return a;
}

static __device__ __forceinline__ float rmax16(float v) {
  v = fmaxf(v, __shfl_xor(v, 1, 32));
  v = fmaxf(v, __shfl_xor(v, 2, 32));
  v = fmaxf(v, __shfl_xor(v, 4, 32));
  v = fmaxf(v, __shfl_xor(v, 8, 32));
  return v;
}
static __device__ __forceinline__ float rsum16(float v) {
  v += __shfl_xor(v, 1, 32);
  v += __shfl_xor(v, 2, 32);
  v += __shfl_xor(v, 4, 32);
  v += __shfl_xor(v, 8, 32);
  return v;
}

// ---- CDNA5 async global->LDS copy (ASYNCcnt-tracked), 16 bytes per enabled lane.
// GV addressing: vdst = LDS byte address, vaddr = 64-bit global address.
static __device__ __forceinline__ void async_copy16(__bf16* lds_dst, const __bf16* gsrc) {
  asm volatile("global_load_async_to_lds_b128 %0, %1, off"
               :: "v"((unsigned int)(size_t)lds_dst),
                  "v"((unsigned long long)(size_t)gsrc)
               : "memory");
}
static __device__ __forceinline__ void wait_async0() {
  asm volatile("s_wait_asynccnt 0x0" ::: "memory");
}

// ---------------- fp32 -> bf16 conversion ----------------
__global__ void cvt_bf16_kernel(const float* __restrict__ src, __bf16* __restrict__ dst, int n) {
  int i = blockIdx.x * blockDim.x + threadIdx.x;
  int stride = gridDim.x * blockDim.x;
  for (; i < n; i += stride) dst[i] = (__bf16)src[i];
}

// ---------------- fused Q/K/V projection ----------------
// One wave computes a 32x16 output strip: each weight B-fragment is reused by two
// A-fragments of x; each A-fragment is reused across the three Q/K/V WMMAs.
// Q stored pre-scaled by 1/sqrt(Hd) into [B,H,S,Hd]; K into [B,H,S,Hd]; V transposed into [B,H,Hd,S].
__global__ void qkv_proj_kernel(const __bf16* __restrict__ xb,
                                const __bf16* __restrict__ wq,
                                const __bf16* __restrict__ wk,
                                const __bf16* __restrict__ wv,
                                const float* __restrict__ bq,
                                const float* __restrict__ bk,
                                const float* __restrict__ bv,
                                __bf16* __restrict__ qo,
                                __bf16* __restrict__ ko,
                                __bf16* __restrict__ vt) {
  const int wid   = blockIdx.x * (blockDim.x >> 5) + (threadIdx.x >> 5);
  const int lane  = threadIdx.x & 31;
  const int col16 = lane & 15;
  const int half  = lane >> 4;
  const int mt = wid >> 6;   // 0..127 : 32-token strip
  const int nt = wid & 63;   // 0..63  : 16-feature tile

  const __bf16* arow0 = xb + (size_t)(mt * 32 + col16) * HIDDEN;
  const __bf16* arow1 = xb + (size_t)(mt * 32 + 16 + col16) * HIDDEN;
  const __bf16* qbrow = wq + (size_t)(nt * 16 + col16) * HIDDEN;   // row e of W (B = W^T)
  const __bf16* kbrow = wk + (size_t)(nt * 16 + col16) * HIDDEN;
  const __bf16* vbrow = wv + (size_t)(nt * 16 + col16) * HIDDEN;

  v8f cq[2], ck[2], cv[2];
#pragma unroll
  for (int u = 0; u < 2; ++u) { cq[u] = (v8f){}; ck[u] = (v8f){}; cv[u] = (v8f){}; }

  for (int k0 = 0; k0 < HIDDEN; k0 += 32) {
    const v16bf a0 = load_frag(arow0, k0, half);
    const v16bf a1 = load_frag(arow1, k0, half);
    const v16bf fq = load_frag(qbrow, k0, half);
    const v16bf fk = load_frag(kbrow, k0, half);
    const v16bf fv = load_frag(vbrow, k0, half);
    cq[0] = WMMA_BF16(a0, fq, cq[0]);
    cq[1] = WMMA_BF16(a1, fq, cq[1]);
    ck[0] = WMMA_BF16(a0, fk, ck[0]);
    ck[1] = WMMA_BF16(a1, fk, ck[1]);
    cv[0] = WMMA_BF16(a0, fv, cv[0]);
    cv[1] = WMMA_BF16(a1, fv, cv[1]);
  }

  const int e  = nt * 16 + col16;
  const int h  = e >> 6;
  const int hd = e & 63;
  const float bqv = bq[e], bkv = bk[e], bvv = bv[e];

#pragma unroll
  for (int u = 0; u < 2; ++u) {
#pragma unroll
    for (int r = 0; r < 8; ++r) {
      const int row = mt * 32 + u * 16 + half * 8 + r;  // 32-row strips never cross batch
      const int b = row >> 11;
      const int s = row & (SEQ - 1);
      const size_t bh = (size_t)(b * HEADS + h);
      qo[(bh * SEQ + s) * HEAD_DIM + hd] = (__bf16)((cq[u][r] + bqv) * 0.125f); // fold 1/sqrt(64)
      ko[(bh * SEQ + s) * HEAD_DIM + hd] = (__bf16)(ck[u][r] + bkv);
      vt[(bh * HEAD_DIM + hd) * SEQ + s] = (__bf16)(cv[u][r] + bvv);
    }
  }
}

// ---------------- flash attention ----------------
// 4 waves per block share one (b,h); each wave owns a 32-query strip.
// K/V tiles for each 32-key chunk are staged into LDS with ASYNCcnt-tracked
// global_load_async_to_lds_b128, double-buffered so the DMA for chunk c+1
// overlaps the 16 WMMAs of chunk c. Fragments are then read from LDS.
__global__ void flash_attn_kernel(const __bf16* __restrict__ q,
                                  const __bf16* __restrict__ k,
                                  const __bf16* __restrict__ vt,
                                  const int* __restrict__ mask,
                                  __bf16* __restrict__ ctx) {
  __shared__ __align__(16) __bf16 kbuf[2][32 * HEAD_DIM];   // 2 x 4KB : K tile [s_local][hd]
  __shared__ __align__(16) __bf16 vbuf[2][HEAD_DIM * 32];   // 2 x 4KB : V tile [hd][s_local]
  __shared__ __align__(16) __bf16 lds_p[4 * 2 * 16 * 32];   // 2KB/wave : two P tiles

  const int tid   = threadIdx.x;         // 0..127
  const int wib   = tid >> 5;
  const int lane  = tid & 31;
  const int col16 = lane & 15;
  const int half  = lane >> 4;
  const int bh = blockIdx.x >> 4;        // 0..31  (all 4 waves share it)
  const int qt = (blockIdx.x & 15) * 4 + wib;  // 0..63 : 32-query strip
  const int b  = bh >> 4;
  const int h  = bh & 15;

  const __bf16* qbase = q  + (size_t)bh * SEQ * HEAD_DIM;
  const __bf16* kbase = k  + (size_t)bh * SEQ * HEAD_DIM;
  const __bf16* vbase = vt + (size_t)bh * HEAD_DIM * SEQ;
  const int*    mrow  = mask + b * SEQ;

  // Q A-fragments for both subtiles, full Hd=64 (two K=32 chunks), resident all loop.
  v16bf aq[2][2];
#pragma unroll
  for (int u = 0; u < 2; ++u) {
    const __bf16* qrow = qbase + (size_t)(qt * 32 + u * 16 + col16) * HEAD_DIM;
    aq[u][0] = load_frag(qrow, 0,  half);
    aq[u][1] = load_frag(qrow, 32, half);
  }

  float mrun[2][8], lrun[2][8];
  v8f acc[2][4];
#pragma unroll
  for (int u = 0; u < 2; ++u) {
#pragma unroll
    for (int r = 0; r < 8; ++r) { mrun[u][r] = -3.0e38f; lrun[u][r] = 0.0f; }
#pragma unroll
    for (int t = 0; t < 4; ++t) acc[u][t] = (v8f){};
  }

  __bf16* pl = &lds_p[wib * 1024];

  // Stage a 32-key chunk: K tile is one contiguous 4KB block; V tile is 64 rows
  // of 64B from the transposed V. 128 threads x 2 copies x 16B each.
  auto stage_kv = [&](int buf, int key0) {
#pragma unroll
    for (int rep = 0; rep < 2; ++rep) {
      const int i = tid + rep * 128;                 // 0..255
      async_copy16(&kbuf[buf][i * 8], kbase + (size_t)key0 * HEAD_DIM + i * 8);
      const int row = i >> 2, part = i & 3;          // 4 x 16B per 32-elem V row
      async_copy16(&vbuf[buf][row * 32 + part * 8],
                   vbase + (size_t)row * SEQ + key0 + part * 8);
    }
  };

  stage_kv(0, 0);
  wait_async0();
  __syncthreads();

  for (int c = 0; c < SEQ / 32; ++c) {
    const int key0 = c * 32;
    const int cur = c & 1;

    // prefetch next chunk while computing this one (previous barrier guarantees
    // the destination buffer is no longer being read)
    if (c + 1 < SEQ / 32) stage_kv(cur ^ 1, key0 + 32);

    // ---- K B-fragments from LDS (shared by both q-subtiles) ----
    const __bf16* kl = kbuf[cur];
    const v16bf bk00 = load_frag(kl + col16 * HEAD_DIM, 0,  half);
    const v16bf bk01 = load_frag(kl + col16 * HEAD_DIM, 32, half);
    const v16bf bk10 = load_frag(kl + (16 + col16) * HEAD_DIM, 0,  half);
    const v16bf bk11 = load_frag(kl + (16 + col16) * HEAD_DIM, 32, half);

    // ---- mask: in C layout each lane owns one key column per subtile ----
    const bool m0 = mrow[key0 + col16] != 0;
    const bool m1 = mrow[key0 + 16 + col16] != 0;

#pragma unroll
    for (int u = 0; u < 2; ++u) {
      // scores: two 16-key subtiles, K-depth 64
      v8f s0 = {}; v8f s1 = {};
      s0 = WMMA_BF16(aq[u][0], bk00, s0);
      s0 = WMMA_BF16(aq[u][1], bk01, s0);
      s1 = WMMA_BF16(aq[u][0], bk10, s1);
      s1 = WMMA_BF16(aq[u][1], bk11, s1);

      // online softmax update
      float scarr[8];
#pragma unroll
      for (int r = 0; r < 8; ++r) {
        const float a0 = m0 ? s0[r] : -3.0e38f;
        const float a1 = m1 ? s1[r] : -3.0e38f;
        const float mx = rmax16(fmaxf(a0, a1));
        const float mnew = fmaxf(mrun[u][r], mx);
        const float sc = __expf(mrun[u][r] - mnew);
        const float e0 = m0 ? __expf(a0 - mnew) : 0.0f;
        const float e1 = m1 ? __expf(a1 - mnew) : 0.0f;
        lrun[u][r] = lrun[u][r] * sc + rsum16(e0 + e1);
        mrun[u][r] = mnew;
        scarr[r] = sc;
        // stage P (bf16) row-major 16x32 in LDS (C layout -> A layout reshuffle)
        pl[u * 512 + (half * 8 + r) * 32 + col16]      = (__bf16)e0;
        pl[u * 512 + (half * 8 + r) * 32 + 16 + col16] = (__bf16)e1;
      }
#pragma unroll
      for (int t = 0; t < 4; ++t)
#pragma unroll
        for (int r = 0; r < 8; ++r) acc[u][t][r] *= scarr[r];
    }

    // ---- read both P tiles back as A-fragments (same-wave LDS, in-order) ----
    v16bf ap[2];
#pragma unroll
    for (int u = 0; u < 2; ++u) {
      const v8bf plo = *reinterpret_cast<const v8bf*>(pl + u * 512 + col16 * 32 + half * 8);
      const v8bf phi = *reinterpret_cast<const v8bf*>(pl + u * 512 + col16 * 32 + 16 + half * 8);
#pragma unroll
      for (int i = 0; i < 8; ++i) { ap[u][i] = plo[i]; ap[u][i + 8] = phi[i]; }
    }

    // ---- acc += P x V : 4 hd tiles from LDS; each V fragment reused by both P tiles ----
    const __bf16* vl = vbuf[cur];
#pragma unroll
    for (int t = 0; t < 4; ++t) {
      const v16bf bv = load_frag(vl + (t * 16 + col16) * 32, 0, half);
      acc[0][t] = WMMA_BF16(ap[0], bv, acc[0][t]);
      acc[1][t] = WMMA_BF16(ap[1], bv, acc[1][t]);
    }

    // publish the prefetched buffer: wait own async copies, then block barrier
    wait_async0();
    __syncthreads();
  }

  // ---- normalize and store ctx (bf16, [B,S,D]) ----
#pragma unroll
  for (int u = 0; u < 2; ++u) {
#pragma unroll
    for (int t = 0; t < 4; ++t) {
#pragma unroll
      for (int r = 0; r < 8; ++r) {
        const int s = qt * 32 + u * 16 + half * 8 + r;
        const float o = acc[u][t][r] / lrun[u][r];
        ctx[(size_t)(b * SEQ + s) * HIDDEN + h * 64 + t * 16 + col16] = (__bf16)o;
      }
    }
  }
}

// ---------------- output projection (fp32 out) ----------------
// 32x16 strip per wave: Wo B-fragment reused by two ctx A-fragments.
__global__ void out_proj_kernel(const __bf16* __restrict__ ctx,
                                const __bf16* __restrict__ wo,
                                const float* __restrict__ bo,
                                float* __restrict__ out) {
  const int wid   = blockIdx.x * (blockDim.x >> 5) + (threadIdx.x >> 5);
  const int lane  = threadIdx.x & 31;
  const int col16 = lane & 15;
  const int half  = lane >> 4;
  const int mt = wid >> 6;   // 0..127
  const int nt = wid & 63;   // 0..63

  const __bf16* arow0 = ctx + (size_t)(mt * 32 + col16) * HIDDEN;
  const __bf16* arow1 = ctx + (size_t)(mt * 32 + 16 + col16) * HIDDEN;
  const __bf16* brow  = wo  + (size_t)(nt * 16 + col16) * HIDDEN;

  v8f c0 = {}; v8f c1 = {};
  for (int k0 = 0; k0 < HIDDEN; k0 += 32) {
    const v16bf a0 = load_frag(arow0, k0, half);
    const v16bf a1 = load_frag(arow1, k0, half);
    const v16bf f  = load_frag(brow,  k0, half);
    c0 = WMMA_BF16(a0, f, c0);
    c1 = WMMA_BF16(a1, f, c1);
  }

  const int e = nt * 16 + col16;
  const float bv = bo[e];
#pragma unroll
  for (int r = 0; r < 8; ++r) {
    const int row0 = mt * 32 + half * 8 + r;
    const int row1 = mt * 32 + 16 + half * 8 + r;
    out[(size_t)row0 * HIDDEN + e] = c0[r] + bv;
    out[(size_t)row1 * HIDDEN + e] = c1[r] + bv;
  }
}

extern "C" void kernel_launch(void* const* d_in, const int* in_sizes, int n_in,
                              void* d_out, int out_size, void* d_ws, size_t ws_size,
                              hipStream_t stream) {
  (void)in_sizes; (void)n_in; (void)out_size; (void)ws_size;

  const float* x    = (const float*)d_in[0];
  const int*   mask = (const int*)  d_in[1];
  const float* Wq   = (const float*)d_in[2];
  const float* bq   = (const float*)d_in[3];
  const float* Wk   = (const float*)d_in[4];
  const float* bk   = (const float*)d_in[5];
  const float* Wv   = (const float*)d_in[6];
  const float* bv   = (const float*)d_in[7];
  const float* Wo   = (const float*)d_in[8];
  const float* bo   = (const float*)d_in[9];
  float* out = (float*)d_out;

  // workspace layout (bf16 elements); total 48 MiB
  __bf16* w    = (__bf16*)d_ws;
  __bf16* xb   = w;                                   // 4096*1024
  __bf16* wqb  = xb  + (size_t)NTOK * HIDDEN;         // 1024*1024
  __bf16* wkb  = wqb + (size_t)HIDDEN * HIDDEN;
  __bf16* wvb  = wkb + (size_t)HIDDEN * HIDDEN;
  __bf16* wob  = wvb + (size_t)HIDDEN * HIDDEN;
  __bf16* qb   = wob + (size_t)HIDDEN * HIDDEN;       // [B,H,S,Hd]
  __bf16* kb   = qb  + (size_t)NTOK * HIDDEN;
  __bf16* vtb  = kb  + (size_t)NTOK * HIDDEN;         // [B,H,Hd,S]
  __bf16* ctxb = vtb + (size_t)NTOK * HIDDEN;         // [B,S,D]

  // fp32 -> bf16 conversions
  cvt_bf16_kernel<<<2048, 256, 0, stream>>>(x,  xb,  NTOK * HIDDEN);
  cvt_bf16_kernel<<<1024, 256, 0, stream>>>(Wq, wqb, HIDDEN * HIDDEN);
  cvt_bf16_kernel<<<1024, 256, 0, stream>>>(Wk, wkb, HIDDEN * HIDDEN);
  cvt_bf16_kernel<<<1024, 256, 0, stream>>>(Wv, wvb, HIDDEN * HIDDEN);
  cvt_bf16_kernel<<<1024, 256, 0, stream>>>(Wo, wob, HIDDEN * HIDDEN);

  // fused QKV projection: 128 x 64 wave-strips = 8192 waves, 8 waves / 256-thread block
  qkv_proj_kernel<<<1024, 256, 0, stream>>>(xb, wqb, wkb, wvb, bq, bk, bv, qb, kb, vtb);

  // flash attention: 32 (b,h) x 16 blocks x 4 wave-strips = 2048 waves
  flash_attn_kernel<<<512, 128, 0, stream>>>(qb, kb, vtb, mask, ctxb);

  // output projection: 8192 waves
  out_proj_kernel<<<1024, 256, 0, stream>>>(ctxb, wob, bo, out);
}
